// FTTLayer_13048110646110
// MI455X (gfx1250) — matrix operand order repaired and checked by reading, not compiled, not run on hardware
//
#include <hip/hip_runtime.h>

typedef float     v2f  __attribute__((ext_vector_type(2)));
typedef float     v8f  __attribute__((ext_vector_type(8)));
typedef _Float16  v4h  __attribute__((ext_vector_type(4)));
typedef _Float16  v8h  __attribute__((ext_vector_type(8)));
typedef _Float16  v16h __attribute__((ext_vector_type(16)));

#define B_TOTAL 32768
#define NDIM    8
#define NNODES  256
#define RANK    32
#define NTHREADS 256

// ---------------- f16-split path ----------------
// K is processed in quarters of 64.  One W "chunk" = one (r, quarter):
// 64 K x 32 S, both hi/lo halves, pre-swizzled to per-lane fragment order.
#define CH_F16   4096                 // f16 elements per chunk (8 KB)
#define NQ       4
#define CH_CFIRST 0
#define CH_CMID   4                   // + (d'*32 + r)*4 + q
#define CH_CLAST  (4 + 6*32*4)        // 772
#define N_CHUNKS  (4 + 6*32*4 + 4)    // 776
#define WS_NEEDED ((size_t)N_CHUNKS * CH_F16 * sizeof(_Float16))

#define BT16     256                  // batch rows per workgroup (f16 path)

// B-matrix (32x16 f16) fragment swizzle: lane = (s&15)+16*((k>>4)&1) holds
// 16 consecutive K as two 8-halves runs (g = (k>>3)&1).
__device__ __forceinline__ int wswzB(int k, int s, int h) {
    int kb = k >> 5, kk = k & 31;
    int hf = (kk >> 4) & 1, g = (kk >> 3) & 1, j = kk & 7;
    int lane = (s & 15) + (hf << 4);
    return ((((kb * 2 + (s >> 4)) * 2 + h) * 2 + g) * 32 + lane) * 8 + j;
}

__global__ __launch_bounds__(NTHREADS)
void split_cores_kernel(const float* __restrict__ cfirst,
                        const float* __restrict__ cmid,
                        const float* __restrict__ clast,
                        _Float16* __restrict__ ws) {
    const int c = blockIdx.x;                 // chunk id 0..775
    _Float16* dst = ws + (size_t)c * CH_F16;
    const int tid = threadIdx.x;
#pragma unroll
    for (int i = 0; i < 8; ++i) {             // 2048 f32 per chunk / 256 thr
        int e = tid + i * NTHREADS;
        int k = e >> 5, s = e & 31;
        float x;
        if (c < CH_CMID) {                    // core_first, q = c
            x = cfirst[(size_t)((c * 64 + k) * 32 + s)];
        } else if (c < CH_CLAST) {            // cores_mid
            int m = c - CH_CMID;
            int q = m & 3, dr = m >> 2;       // dr = d'*32 + r
            x = cmid[((size_t)dr * NNODES + q * 64 + k) * 32 + s];
        } else {                              // core_last, TRANSPOSED: W[k][s]=clast[s,.,k]
            int q = c - CH_CLAST;
            x = clast[(size_t)s * NNODES + q * 64 + k];
        }
        _Float16 hi = (_Float16)x;
        _Float16 lo = (_Float16)(x - (float)hi);
        dst[wswzB(k, s, 0)] = hi;
        dst[wswzB(k, s, 1)] = lo;
    }
}

__device__ __forceinline__ v16h cat8(v8h a, v8h b) {
    return __builtin_shufflevector(a, b, 0,1,2,3,4,5,6,7,8,9,10,11,12,13,14,15);
}

__global__ __launch_bounds__(NTHREADS)
void ftt_f16split_kernel(const float* __restrict__ bf,      // (B,8,256) f32
                         const _Float16* __restrict__ wsc,  // pre-split cores
                         float* __restrict__ out)           // (B,1)
{
    __shared__ _Float16 bfLds[32768];      // 64 KB: [kb][h][g][half][row256][8]
    __shared__ float    wbuf[2 * 2048];    // 16 KB: double-buffered chunk
    __shared__ float    vbuf[2 * BT16 * RANK]; // 64 KB: v ping-pong

    const int tid     = threadIdx.x;
    const int lane    = tid & 31;
    const int lane_lo = lane & 15;
    const int half    = lane >> 4;
    const int wv      = tid >> 5;
    const int mrow0[2] = { wv * 16, wv * 16 + 128 };  // 2 M-tiles per wave
    const size_t b0   = (size_t)blockIdx.x * BT16;

    // stage one K-quarter of bf, splitting f32 -> f16 hi/lo, swizzled for A frags
    auto stage_bf = [&](int d, int nq) {
        const float* src = bf + b0 * (NDIM * NNODES) + d * NNODES + nq * 64;
#pragma unroll
        for (int p = 0; p < 16; ++p) {
            int e4  = tid + p * NTHREADS;          // float4 id (4096 total)
            int row = e4 >> 4;
            int k0  = (e4 & 15) * 4;
            float4 x = *(const float4*)(src + (size_t)row * (NDIM * NNODES) + k0);
            _Float16 h0 = (_Float16)x.x, h1 = (_Float16)x.y,
                     h2 = (_Float16)x.z, h3 = (_Float16)x.w;
            _Float16 l0 = (_Float16)(x.x - (float)h0), l1 = (_Float16)(x.y - (float)h1),
                     l2 = (_Float16)(x.z - (float)h2), l3 = (_Float16)(x.w - (float)h3);
            int kb = k0 >> 5, r5 = k0 & 31;
            int g = (r5 >> 3) & 1, hf = (r5 >> 4) & 1, j = r5 & 7;
            int offH = ((((kb * 2 + 0) * 2 + g) * 2 + hf) * 256 + row) * 8 + j;
            int offL = ((((kb * 2 + 1) * 2 + g) * 2 + hf) * 256 + row) * 8 + j;
            *(v4h*)(bfLds + offH) = (v4h){h0, h1, h2, h3};
            *(v4h*)(bfLds + offL) = (v4h){l0, l1, l2, l3};
        }
    };
    // CDNA5 async DMA: copy one pre-swizzled 8 KB chunk global -> LDS, no VGPR
    // round-trip.  GVS mode: LDS-address VGPR, 32-bit offset VGPR, SGPR64 base.
    // Tracked on ASYNCcnt; published with s_wait_asynccnt + barrier.
    auto stage_w_async = [&](float* dstW, const _Float16* chunk) {
        unsigned ldsa = (unsigned)(uintptr_t)dstW + (unsigned)tid * 32u;
        unsigned voff = (unsigned)tid * 32u;
        asm volatile("global_load_async_to_lds_b128 %0, %1, %2 offset:0\n\t"
                     "global_load_async_to_lds_b128 %0, %1, %2 offset:16"
                     :: "v"(ldsa), "v"(voff), "s"(chunk) : "memory");
    };
    auto wait_async = [&]() {
        asm volatile("s_wait_asynccnt 0x0" ::: "memory");
    };
    // A fragments for this quarter, register-resident across the whole r loop
    v16h Af[2][2][2];   // [mtile][kb][hi/lo]
    auto load_afrags = [&]() {
#pragma unroll
        for (int mt = 0; mt < 2; ++mt)
#pragma unroll
            for (int kb = 0; kb < 2; ++kb)
#pragma unroll
                for (int h = 0; h < 2; ++h) {
                    int grp = ((kb * 2 + h) * 2 + 0) * 2 + half;   // g = 0
                    const _Float16* pa = bfLds + grp * 2048 + (mrow0[mt] + lane_lo) * 8;
                    Af[mt][kb][h] = cat8(*(const v8h*)pa, *(const v8h*)(pa + 4096));
                }
    };
    auto load_bfrag = [&](const _Float16* wl, int kb, int t, int h) -> v16h {
        int grp = ((kb * 2 + t) * 2 + h) * 2;                      // g = 0
        const _Float16* pb = wl + grp * 256 + lane * 8;
        return cat8(*(const v8h*)pb, *(const v8h*)(pb + 256));
    };

    v8f C[2][2];
#pragma unroll
    for (int mt = 0; mt < 2; ++mt) { C[mt][0] = (v8f){}; C[mt][1] = (v8f){}; }

    // one chunk of GEMM: 3-WMMA error-compensated f16 (AhiBhi + AhiBlo + AloBhi)
    auto gemm_chunk = [&](const _Float16* wl, v8f acc[2][2]) {
#pragma unroll
        for (int kb = 0; kb < 2; ++kb) {
            v16h Bh0 = load_bfrag(wl, kb, 0, 0);
            v16h Bl0 = load_bfrag(wl, kb, 0, 1);
            v16h Bh1 = load_bfrag(wl, kb, 1, 0);
            v16h Bl1 = load_bfrag(wl, kb, 1, 1);
#pragma unroll
            for (int mt = 0; mt < 2; ++mt) {
                v16h Ah = Af[mt][kb][0], Al = Af[mt][kb][1];
                acc[mt][0] = __builtin_amdgcn_wmma_f32_16x16x32_f16(false, Ah, false, Bh0, (short)0, acc[mt][0], false, false);
                acc[mt][1] = __builtin_amdgcn_wmma_f32_16x16x32_f16(false, Ah, false, Bh1, (short)0, acc[mt][1], false, false);
                acc[mt][0] = __builtin_amdgcn_wmma_f32_16x16x32_f16(false, Ah, false, Bl0, (short)0, acc[mt][0], false, false);
                acc[mt][1] = __builtin_amdgcn_wmma_f32_16x16x32_f16(false, Ah, false, Bl1, (short)0, acc[mt][1], false, false);
                acc[mt][0] = __builtin_amdgcn_wmma_f32_16x16x32_f16(false, Al, false, Bh0, (short)0, acc[mt][0], false, false);
                acc[mt][1] = __builtin_amdgcn_wmma_f32_16x16x32_f16(false, Al, false, Bh1, (short)0, acc[mt][1], false, false);
            }
        }
    };
    auto store_acc = [&](float* vdst) {
#pragma unroll
        for (int mt = 0; mt < 2; ++mt)
#pragma unroll
            for (int i = 0; i < 8; ++i) {
                int row = mrow0[mt] + i + 8 * half;
                vdst[row * RANK + lane_lo]      = C[mt][0][i];
                vdst[row * RANK + lane_lo + 16] = C[mt][1][i];
            }
    };

    // ---------- step 0: v = bf0 @ core_first (direct accumulate) ----------
    for (int nq = 0; nq < NQ; ++nq) {
        __syncthreads();
        stage_w_async(wbuf, wsc + (size_t)(CH_CFIRST + nq) * CH_F16);
        stage_bf(0, nq);
        wait_async();
        __syncthreads();
        load_afrags();
        gemm_chunk((const _Float16*)wbuf, C);
    }
    store_acc(vbuf);
    int cur = 0;

    // ---------- steps 1..6: C[m,s] += sum_r v[m,r] * (bf_d @ core_d[r]) ----------
    for (int d = 1; d <= NDIM - 2; ++d) {
#pragma unroll
        for (int mt = 0; mt < 2; ++mt) { C[mt][0] = (v8f){}; C[mt][1] = (v8f){}; }
        const float* vcur = vbuf + cur * BT16 * RANK;
        const int chbase = CH_CMID + (d - 1) * RANK * NQ;
        for (int nq = 0; nq < NQ; ++nq) {
            __syncthreads();
            stage_w_async(wbuf, wsc + (size_t)(chbase + nq) * CH_F16);   // r = 0
            stage_bf(d, nq);
            wait_async();
            __syncthreads();
            load_afrags();
            for (int r = 0; r < RANK; ++r) {
                const bool more = (r + 1 < RANK);
                if (more)                                  // async prefetch next chunk
                    stage_w_async(wbuf + ((r + 1) & 1) * 2048,
                                  wsc + (size_t)(chbase + (r + 1) * NQ + nq) * CH_F16);
                v8f T[2][2];
#pragma unroll
                for (int mt = 0; mt < 2; ++mt) { T[mt][0] = (v8f){}; T[mt][1] = (v8f){}; }
                gemm_chunk((const _Float16*)(wbuf + (r & 1) * 2048), T);
#pragma unroll
                for (int mt = 0; mt < 2; ++mt) {
                    const float* vp = vcur + (mrow0[mt] + 8 * half) * RANK + r;
#pragma unroll
                    for (int i = 0; i < 8; ++i) {
                        float vr = vp[i * RANK];          // broadcast LDS read
                        C[mt][0][i] += vr * T[mt][0][i];
                        C[mt][1][i] += vr * T[mt][1][i];
                    }
                }
                if (more) wait_async();
                __syncthreads();
            }
        }
        store_acc(vbuf + (cur ^ 1) * BT16 * RANK);
        cur ^= 1;
    }

    // ---------- step 7: t = bf7 @ core_last^T ; out = sum_r v*t ----------
#pragma unroll
    for (int mt = 0; mt < 2; ++mt) { C[mt][0] = (v8f){}; C[mt][1] = (v8f){}; }
    for (int nq = 0; nq < NQ; ++nq) {
        __syncthreads();
        stage_w_async(wbuf, wsc + (size_t)(CH_CLAST + nq) * CH_F16);
        stage_bf(NDIM - 1, nq);
        wait_async();
        __syncthreads();
        load_afrags();
        gemm_chunk((const _Float16*)wbuf, C);
    }
    float* tbuf = vbuf + (cur ^ 1) * BT16 * RANK;
    store_acc(tbuf);
    __syncthreads();
    {
        const float* vrow = vbuf + cur * BT16 * RANK + tid * RANK;
        const float* trow = tbuf + tid * RANK;
        float sum = 0.f;
#pragma unroll
        for (int r = 0; r < RANK; ++r) sum += vrow[r] * trow[r];
        out[b0 + tid] = sum;
    }
}

// ================= f32 WMMA fallback (used only if ws too small) =================
#define BT      128
#define BF_STRIDE 132
#define V_STRIDE  33
#define WCHUNK  4096

__device__ __forceinline__ int wswz(int k, int s) {
    return ((((k >> 2) * 2 + (s >> 4)) << 5) + ((k & 2) << 3) + (s & 15)) * 2 + (k & 1);
}
__device__ __forceinline__ void gemm_chunk32(const float* wchunk, const float* bf_row,
                                             float vr, int lane, v8f& acc0, v8f& acc1) {
    const v2f* wp = (const v2f*)wchunk;
#pragma unroll 8
    for (int n0 = 0; n0 < 32; ++n0) {
        v2f bfp = *(const v2f*)(bf_row + n0 * 4);
        v2f a;  a.x = vr * bfp.x;  a.y = vr * bfp.y;
        v2f w0 = wp[n0 * 64 + lane];
        v2f w1 = wp[n0 * 64 + 32 + lane];
        acc0 = __builtin_amdgcn_wmma_f32_16x16x4_f32(false, a, false, w0, (short)0, acc0, false, false);
        acc1 = __builtin_amdgcn_wmma_f32_16x16x4_f32(false, a, false, w1, (short)0, acc1, false, false);
    }
}
__global__ __launch_bounds__(NTHREADS)
void ftt_f32_kernel(const float* __restrict__ bf, const float* __restrict__ cfirst,
                    const float* __restrict__ cmid, const float* __restrict__ clast,
                    float* __restrict__ out) {
    __shared__ float bfT[BT * BF_STRIDE];
    __shared__ float vbuf[2 * BT * V_STRIDE];
    __shared__ float wbuf[2 * WCHUNK];
    const int tid = threadIdx.x, lane = tid & 31, lane_lo = lane & 15, half = lane >> 4;
    const int m0 = (tid >> 5) * 16, b0 = blockIdx.x * BT;
    const float* bf_row = bfT + (m0 + lane_lo) * BF_STRIDE + 2 * half;
    auto stage_bf = [&](int d, int nh) {
        const float* src = bf + (size_t)b0 * (NDIM * NNODES) + d * NNODES + nh * 128;
#pragma unroll
        for (int q = 0; q < 16; ++q) {
            int e = tid + q * NTHREADS, row = e >> 5, c4 = e & 31;
            float4 v = *(const float4*)(src + (size_t)row * (NDIM * NNODES) + c4 * 4);
            *(float4*)&bfT[row * BF_STRIDE + c4 * 4] = v;
        }
    };
    auto stage_w = [&](float* dst, const float* src) {
#pragma unroll
        for (int q = 0; q < 4; ++q) {
            int e = tid * 16 + q * 4;
            float4 v = *(const float4*)(src + e);
            int k = e >> 5, s = e & 31;
            dst[wswz(k, s + 0)] = v.x;  dst[wswz(k, s + 1)] = v.y;
            dst[wswz(k, s + 2)] = v.z;  dst[wswz(k, s + 3)] = v.w;
        }
    };
    auto store_acc = [&](float* vdst, const v8f& a0, const v8f& a1) {
#pragma unroll
        for (int i = 0; i < 8; ++i) {
            int row = m0 + i + 8 * half;
            vdst[row * V_STRIDE + lane_lo]      = a0[i];
            vdst[row * V_STRIDE + lane_lo + 16] = a1[i];
        }
    };
    v8f acc0 = {}, acc1 = {};
    for (int nh = 0; nh < 2; ++nh) {
        __syncthreads(); stage_bf(0, nh); stage_w(wbuf, cfirst + nh * WCHUNK); __syncthreads();
        gemm_chunk32(wbuf, bf_row, 1.0f, lane, acc0, acc1);
    }
    store_acc(vbuf, acc0, acc1);
    int cur = 0;
    for (int d = 1; d <= NDIM - 2; ++d) {
        acc0 = {}; acc1 = {};
        const float* vcur = vbuf + cur * BT * V_STRIDE;
        const float* dbase = cmid + (size_t)(d - 1) * RANK * NNODES * RANK;
        for (int nh = 0; nh < 2; ++nh) {
            __syncthreads(); stage_bf(d, nh); stage_w(wbuf, dbase + nh * WCHUNK); __syncthreads();
            for (int r = 0; r < RANK; ++r) {
                float4 pre[4];
                const bool more = (r + 1 < RANK);
                if (more) {
                    const float* nsrc = dbase + (size_t)(r + 1) * (NNODES * RANK) + nh * WCHUNK;
#pragma unroll
                    for (int q = 0; q < 4; ++q) pre[q] = *(const float4*)(nsrc + tid * 16 + q * 4);
                }
                const float vr = vcur[(m0 + lane_lo) * V_STRIDE + r];
                gemm_chunk32(wbuf + (r & 1) * WCHUNK, bf_row, vr, lane, acc0, acc1);
                if (more) {
                    float* dst = wbuf + ((r + 1) & 1) * WCHUNK;
#pragma unroll
                    for (int q = 0; q < 4; ++q) {
                        int e = tid * 16 + q * 4, k = e >> 5, s = e & 31;
                        dst[wswz(k, s + 0)] = pre[q].x;  dst[wswz(k, s + 1)] = pre[q].y;
                        dst[wswz(k, s + 2)] = pre[q].z;  dst[wswz(k, s + 3)] = pre[q].w;
                    }
                }
                __syncthreads();
            }
        }
        store_acc(vbuf + (cur ^ 1) * BT * V_STRIDE, acc0, acc1);
        cur ^= 1;
    }
    acc0 = {}; acc1 = {};
    for (int nh = 0; nh < 2; ++nh) {
        __syncthreads(); stage_bf(NDIM - 1, nh);
#pragma unroll
        for (int q = 0; q < 4; ++q) {
            int e = tid * 16 + q * 4, k = e >> 5, s = e & 31;
#pragma unroll
            for (int j = 0; j < 4; ++j)
                wbuf[wswz(k, s + j)] = clast[(size_t)(s + j) * NNODES + nh * 128 + k];
        }
        __syncthreads();
        gemm_chunk32(wbuf, bf_row, 1.0f, lane, acc0, acc1);
    }
    float* tbuf = vbuf + (cur ^ 1) * BT * V_STRIDE;
    store_acc(tbuf, acc0, acc1);
    __syncthreads();
    if (tid < BT) {
        const float* vrow = vbuf + cur * BT * V_STRIDE + tid * V_STRIDE;
        const float* trow = tbuf + tid * V_STRIDE;
        float sum = 0.f;
#pragma unroll
        for (int r = 0; r < RANK; ++r) sum += vrow[r] * trow[r];
        out[b0 + tid] = sum;
    }
}

extern "C" void kernel_launch(void* const* d_in, const int* in_sizes, int n_in,
                              void* d_out, int out_size, void* d_ws, size_t ws_size,
                              hipStream_t stream) {
    const float* bf     = (const float*)d_in[0];
    const float* cfirst = (const float*)d_in[1];
    const float* cmid   = (const float*)d_in[2];
    const float* clast  = (const float*)d_in[3];
    float* out = (float*)d_out;
    if (ws_size >= WS_NEEDED) {
        _Float16* ws = (_Float16*)d_ws;
        split_cores_kernel<<<dim3(N_CHUNKS), dim3(NTHREADS), 0, stream>>>(cfirst, cmid, clast, ws);
        ftt_f16split_kernel<<<dim3(B_TOTAL / BT16), dim3(NTHREADS), 0, stream>>>(bf, ws, out);
    } else {
        ftt_f32_kernel<<<dim3(B_TOTAL / BT), dim3(NTHREADS), 0, stream>>>(bf, cfirst, cmid, clast, out);
    }
}
